// HAN_5188320494365
// MI455X (gfx1250) — matrix operand order repaired and checked by reading, not compiled, not run on hardware
//
#include <hip/hip_runtime.h>
#include <hip/hip_bf16.h>

// ---------------------------------------------------------------------------
// Problem constants (match the reference)
// ---------------------------------------------------------------------------
#define NE_    100000
#define NREG_  2000
#define D_     128
#define NR_    8
#define NMP_   3
#define NRS_   4
#define NSUB_  40000
#define EG_    1600000
#define EM_    400000
#define PRE_   768
#define NTASK_ 2
#define SUM_L_ 2
#define SUB_L_ 2

#define ACT_NONE 0
#define ACT_TANH 1
#define ACT_RELU 2

#define BM_ 64    // block tile M
#define BN_ 128   // block tile N (full width -> grid.y == 1, A read once)
// 4 waves arranged 2(M) x 2(N); wave tile 32(M) x 64(N) -> 8 v8f accumulators
// (64 VGPRs) per wave: the spill-free configuration.

typedef _Float16 half8 __attribute__((ext_vector_type(8)));
typedef _Float16 v16h  __attribute__((ext_vector_type(16)));
typedef float    v8f   __attribute__((ext_vector_type(8)));

__device__ __forceinline__ v16h cat16(half8 lo, half8 hi) {
  return __builtin_shufflevector(lo, hi, 0,1,2,3,4,5,6,7,8,9,10,11,12,13,14,15);
}

__device__ __forceinline__ void atomic_add_f32(float* p, float v) {
  __hip_atomic_fetch_add(p, v, __ATOMIC_RELAXED, __HIP_MEMORY_SCOPE_AGENT);
}

// ---------------------------------------------------------------------------
// WMMA GEMM:  C[M,N] = act( scaleA(A)[M,K] @ B[K,N] + bias (+ C if accum) )
//
// If cntA != nullptr, A is a per-(row, K/128-block) segment-sum buffer and each
// staged element is scaled by 1/cnt (0 if cnt==0) -- fuses the RGCN "mean"
// step into the GEMM, avoiding a full read+write pass over A.
//
// f16 compute / f32 accumulate via v_wmma_f32_16x16x32_f16.
// 128 threads = 4 waves; block tile 64x128; wave tile 32(M)x64(N):
// 2 A frags x 4 B frags = 8 WMMAs per 32-wide K step, each B frag reused 2x.
// LDS double-buffered (24 KB): stage tile k+1 while computing tile k.
// K % 32 == 0, N == 128 here.
// ---------------------------------------------------------------------------
__global__ __launch_bounds__(128)
void han_gemm_wmma(const float* __restrict__ A, const float* __restrict__ B,
                   const float* __restrict__ bias, float* __restrict__ C,
                   const float* __restrict__ cntA, int RA,
                   int M, int N, int K, int accum, int act)
{
  __shared__ __align__(16) _Float16 As[2][BM_][32];   // [buf][m_local][k_local]
  __shared__ __align__(16) _Float16 Bs[2][BN_][32];   // [buf][n_local][k_local]

  const int m0   = blockIdx.x * BM_;
  const int n0   = blockIdx.y * BN_;
  const int tid  = threadIdx.x;
  const int lane = tid & 31;
  const int wv   = tid >> 5;
  const int mw   = wv & 1;        // wave's M half of the block
  const int nw   = wv >> 1;       // wave's N half of the block
  const int hs   = lane >> 4;     // half-wave select
  const int l16  = lane & 15;
  const bool interior = (m0 + BM_ <= M);

  v8f acc[2][4];
  for (int f = 0; f < 2; ++f)
    for (int a = 0; a < 4; ++a) {
      if (accum) {
        const int col = n0 + nw * 64 + a * 16 + l16;
        for (int v = 0; v < 8; ++v) {
          const int m = m0 + mw * 32 + f * 16 + hs * 8 + v;
          acc[f][a][v] = (m < M) ? C[(long)m * N + col] : 0.0f;
        }
      } else {
        for (int v = 0; v < 8; ++v) acc[f][a][v] = 0.0f;
      }
    }

  // Stage one 32-wide K tile (A: BM_x32, B transposed: BN_x32) into LDS buf.
  auto stage = [&](int buf, int ks) {
    if (interior) {
      // fast path: no M guards; BM_*8 float4 / 128 threads = 4 each
      for (int idx = tid; idx < BM_ * 8; idx += 128) {
        const int r = idx >> 3, c4 = idx & 7;
        const int m = m0 + r;
        float4 v = *(const float4*)(A + (long)m * K + ks + c4 * 4);
        if (cntA) {
          const float c = cntA[(long)m * RA + (ks >> 7)];
          const float inv = (c > 0.0f) ? (1.0f / c) : 0.0f;
          v.x *= inv; v.y *= inv; v.z *= inv; v.w *= inv;
        }
        _Float16* p = &As[buf][r][c4 * 4];
        p[0] = (_Float16)v.x; p[1] = (_Float16)v.y;
        p[2] = (_Float16)v.z; p[3] = (_Float16)v.w;
      }
      // prefetch next K tile of A (global_prefetch_b8 path)
      if (ks + 32 < K && tid < BM_)
        __builtin_prefetch(A + (long)(m0 + tid) * K + ks + 32, 0, 0);
    } else {
      for (int idx = tid; idx < BM_ * 8; idx += 128) {
        const int r = idx >> 3, c4 = idx & 7;
        const int m = m0 + r;
        float4 v = make_float4(0.f, 0.f, 0.f, 0.f);
        if (m < M) {
          v = *(const float4*)(A + (long)m * K + ks + c4 * 4);
          if (cntA) {
            const float c = cntA[(long)m * RA + (ks >> 7)];
            const float inv = (c > 0.0f) ? (1.0f / c) : 0.0f;
            v.x *= inv; v.y *= inv; v.z *= inv; v.w *= inv;
          }
        }
        _Float16* p = &As[buf][r][c4 * 4];
        p[0] = (_Float16)v.x; p[1] = (_Float16)v.y;
        p[2] = (_Float16)v.z; p[3] = (_Float16)v.w;
      }
    }
    // B tile transposed: 32(K) x BN_(N) f32 -> Bs[n][k] f16; always in range.
    for (int idx = tid; idx < BN_ * 8; idx += 128) {
      const int kk = idx >> 5, c4 = idx & 31;       // 32 float4 per K row
      const float4 v = *(const float4*)(B + (long)(ks + kk) * N + n0 + c4 * 4);
      Bs[buf][c4 * 4 + 0][kk] = (_Float16)v.x;
      Bs[buf][c4 * 4 + 1][kk] = (_Float16)v.y;
      Bs[buf][c4 * 4 + 2][kk] = (_Float16)v.z;
      Bs[buf][c4 * 4 + 3][kk] = (_Float16)v.w;
    }
  };

  // Consume one staged tile: 2 A frags x 4 B frags -> 8 WMMAs.
  auto compute = [&](int buf) {
    // A fragment (ISA 7.12.2): lane row = l16, elems 0..7 = K kb..kb+7,
    // elems 8..15 = K kb+16..kb+23, kb = hs*8.
    const _Float16* ap0 = &As[buf][mw * 32 + l16][hs * 8];
    const _Float16* ap1 = &As[buf][mw * 32 + 16 + l16][hs * 8];
    const v16h af0 = cat16(*(const half8*)ap0, *(const half8*)(ap0 + 16));
    const v16h af1 = cat16(*(const half8*)ap1, *(const half8*)(ap1 + 16));
    for (int a = 0; a < 4; ++a) {
      // B fragment: lane col = l16, elems 0..7 = K hs*8.., elems 8..15 = K 16+hs*8...
      const _Float16* bp = &Bs[buf][nw * 64 + a * 16 + l16][hs * 8];
      const v16h bf = cat16(*(const half8*)bp, *(const half8*)(bp + 16));
      acc[0][a] = __builtin_amdgcn_wmma_f32_16x16x32_f16(
          false, af0, false, bf, (short)0, acc[0][a], false, false);
      acc[1][a] = __builtin_amdgcn_wmma_f32_16x16x32_f16(
          false, af1, false, bf, (short)0, acc[1][a], false, false);
    }
  };

  stage(0, 0);
  int buf = 0;
  for (int k0 = 0; k0 < K; k0 += 32) {
    __syncthreads();
    if (k0 + 32 < K) stage(buf ^ 1, k0 + 32);
    compute(buf);
    buf ^= 1;
  }

  for (int f = 0; f < 2; ++f)
    for (int a = 0; a < 4; ++a) {
      const int col  = n0 + nw * 64 + a * 16 + l16;
      const float bv = bias ? bias[col] : 0.0f;
      for (int v = 0; v < 8; ++v) {
        const int m = m0 + mw * 32 + f * 16 + hs * 8 + v;
        if (m < M) {
          float r = acc[f][a][v] + bv;
          if      (act == ACT_TANH) r = tanhf(r);
          else if (act == ACT_RELU) r = fmaxf(r, 0.0f);
          C[(long)m * N + col] = r;
        }
      }
    }
}

// ---------------------------------------------------------------------------
// Edge scatter: one wave per edge; lane i handles float4 i of the 128-float row.
// sums[(dst*R+etype)*128 + :] += x[src,:]; cnt[dst*R+etype] += 1
// ---------------------------------------------------------------------------
__global__ __launch_bounds__(256)
void han_scatter(const float* __restrict__ x, const int* __restrict__ src,
                 const int* __restrict__ dst, const int* __restrict__ et,
                 float* __restrict__ sums, float* __restrict__ cnt, int E, int R)
{
  const int wave = (int)((blockIdx.x * 256 + threadIdx.x) >> 5);
  const int lane = threadIdx.x & 31;
  if (wave >= E) return;
  const int s = src[wave], d = dst[wave], t = et[wave];
  const float4 v = *((const float4*)(x + (long)s * D_) + lane);
  float* o = sums + ((long)d * R + t) * D_ + lane * 4;
  atomic_add_f32(o + 0, v.x);
  atomic_add_f32(o + 1, v.y);
  atomic_add_f32(o + 2, v.z);
  atomic_add_f32(o + 3, v.w);
  if (lane == 0) atomic_add_f32(cnt + (long)d * R + t, 1.0f);
}

// ---------------------------------------------------------------------------
// Gather through the CDNA5 async-LDS DMA path:
//   feat[i,:] = E[ids[i],:]
// Each thread copies one float4 via
//   GLOBAL_LOAD_ASYNC_TO_LDS_B128 -> S_WAIT_ASYNCCNT -> GLOBAL_STORE_ASYNC_FROM_LDS_B128
// so the row never round-trips through VGPRs. The LDS operand is the
// wave-relative LDS offset (low 32 bits of the generic pointer, ISA 10.2).
// ---------------------------------------------------------------------------
__global__ __launch_bounds__(256)
void han_gather_async(const float* __restrict__ E, const int* __restrict__ ids,
                      float* __restrict__ out, int n)
{
  __shared__ __align__(16) float buf[256 * 4];
  const int i = blockIdx.x * 256 + threadIdx.x;   // float4 index over n*32
  const unsigned lds = (unsigned)(uintptr_t)(&buf[threadIdx.x * 4]);
  if (i < n * 32) {
    const int node = i >> 5;
    const unsigned long long gsrc =
        (unsigned long long)(uintptr_t)(E + (long)ids[node] * D_ + (i & 31) * 4);
    const unsigned long long gdst =
        (unsigned long long)(uintptr_t)(out + (long)i * 4);
    asm volatile("global_load_async_to_lds_b128 %0, %1, off"
                 :: "v"(lds), "v"(gsrc) : "memory");
    asm volatile("s_wait_asynccnt 0x0" ::: "memory");
    asm volatile("global_store_async_from_lds_b128 %0, %1, off"
                 :: "v"(gdst), "v"(lds) : "memory");
    asm volatile("s_wait_asynccnt 0x0" ::: "memory");
  }
}

// Tiny attention: scores_i = tanh(emb_i @ W1 + b1) . W2 ; wout = softmax(scores)
__global__ __launch_bounds__(128)
void han_attn(const float* __restrict__ emb, const float* __restrict__ W1,
              const float* __restrict__ b1, const float* __restrict__ W2,
              float* __restrict__ wout, int n, int Kdim)
{
  __shared__ float red[128];
  __shared__ float sc[8];
  const int d = threadIdx.x;
  for (int i = 0; i < n; ++i) {
    float acc = b1[d];
    for (int k = 0; k < Kdim; ++k)
      acc = fmaf(emb[i * Kdim + k], W1[k * 128 + d], acc);
    red[d] = tanhf(acc) * W2[d];
    __syncthreads();
    for (int s = 64; s > 0; s >>= 1) {
      if (d < s) red[d] += red[d + s];
      __syncthreads();
    }
    if (d == 0) sc[i] = red[0];
    __syncthreads();
  }
  if (d == 0) {
    float mx = sc[0];
    for (int i = 1; i < n; ++i) mx = fmaxf(mx, sc[i]);
    float e[8], s = 0.0f;
    for (int i = 0; i < n; ++i) { e[i] = expf(sc[i] - mx); s += e[i]; }
    for (int i = 0; i < n; ++i) wout[i] = e[i] / s;
  }
}

// h = w0*sem0 + w1*sem1 + w2*sem2   (float4 granularity)
__global__ void han_semsum(const float4* __restrict__ s0, const float4* __restrict__ s1,
                           const float4* __restrict__ s2, const float* __restrict__ w,
                           float4* __restrict__ h, int total4)
{
  const int i = blockIdx.x * blockDim.x + threadIdx.x;
  if (i < total4) {
    const float w0 = w[0], w1 = w[1], w2 = w[2];
    const float4 a = s0[i], b = s1[i], c = s2[i];
    float4 r;
    r.x = w0 * a.x + w1 * b.x + w2 * c.x;
    r.y = w0 * a.y + w1 * b.y + w2 * c.y;
    r.z = w0 * a.z + w1 * b.z + w2 * c.z;
    r.w = w0 * a.w + w1 * b.w + w2 * c.w;
    h[i] = r;
  }
}

// out[n] = ((1+wt0)*E_reg + wt1*(o_reg+o_kg) + E_glob) . Wer + ber  (wave per row)
__global__ __launch_bounds__(256)
void han_final(const float4* __restrict__ Ereg, const float4* __restrict__ oreg,
               const float4* __restrict__ okg, const float4* __restrict__ Efin,
               const float4* __restrict__ Wer, const float* __restrict__ ber,
               const float* __restrict__ wt, float* __restrict__ out, int n)
{
  const int wave = (int)((blockIdx.x * 256 + threadIdx.x) >> 5);
  const int lane = threadIdx.x & 31;
  if (wave >= n) return;
  const float c0 = 1.0f + wt[0], c1 = wt[1];
  const long base = (long)wave * 32 + lane;
  const float4 e = Ereg[base], o = oreg[base], k = okg[base], g = Efin[base];
  const float4 wr = Wer[lane];
  float s = (e.x * c0 + c1 * (o.x + k.x) + g.x) * wr.x
          + (e.y * c0 + c1 * (o.y + k.y) + g.y) * wr.y
          + (e.z * c0 + c1 * (o.z + k.z) + g.z) * wr.z
          + (e.w * c0 + c1 * (o.w + k.w) + g.w) * wr.w;
  for (int off = 16; off; off >>= 1) s += __shfl_xor(s, off, 32);
  if (lane == 0) out[wave] = s + ber[0];
}

// ---------------------------------------------------------------------------
// Host-side orchestration
// ---------------------------------------------------------------------------
extern "C" void kernel_launch(void* const* d_in, const int* in_sizes, int n_in,
                              void* d_out, int out_size, void* d_ws, size_t ws_size,
                              hipStream_t stream)
{
  (void)in_sizes; (void)n_in; (void)out_size; (void)ws_size;

  const float* E_pre      = (const float*)d_in[0];
  const float* W_map      = (const float*)d_in[1];
  const float* b_map      = (const float*)d_in[2];
  const float* rgcn_Wrel  = (const float*)d_in[3];   // [2,8,128,128]
  const float* rgcn_Wroot = (const float*)d_in[4];   // [2,128,128]
  const float* rgcn_b     = (const float*)d_in[5];   // [2,128]
  const float* han_Wrel   = (const float*)d_in[6];   // [3,2,4,128,128]
  const float* han_Wroot  = (const float*)d_in[7];   // [3,2,128,128]
  const float* han_b      = (const float*)d_in[8];   // [3,2,128]
  const float* mp_emb     = (const float*)d_in[9];   // [3,768]
  const float* Wp1        = (const float*)d_in[10];
  const float* bp1        = (const float*)d_in[11];
  const float* Wp2        = (const float*)d_in[12];
  const float* Wpred      = (const float*)d_in[13];
  const float* bpred      = (const float*)d_in[14];
  const float* task_emb   = (const float*)d_in[15];
  const float* Wt1        = (const float*)d_in[16];
  const float* bt1        = (const float*)d_in[17];
  const float* Wt2        = (const float*)d_in[18];
  const float* other_reg  = (const float*)d_in[19];
  const float* other_kg   = (const float*)d_in[20];
  const float* Wer        = (const float*)d_in[21];
  const float* ber        = (const float*)d_in[22];
  const int*   edge_index = (const int*)d_in[23];    // [2,EG]
  const int*   edge_type  = (const int*)d_in[24];
  const int*   mp_ei      = (const int*)d_in[25];    // [3,2,EM]
  const int*   mp_et      = (const int*)d_in[26];    // [3,EM]
  const int*   mp_eids    = (const int*)d_in[27];    // [3,NSUB]

  // Workspace bump allocator (256B aligned slabs).
  char* base = (char*)d_ws;
  size_t off = 0;
  auto alloc = [&](size_t bytes) -> float* {
    float* p = (float*)(base + off);
    off += (bytes + 255) & ~(size_t)255;
    return p;
  };
  float* E_a    = alloc((size_t)NE_ * D_ * 4);
  float* E_b    = alloc((size_t)NE_ * D_ * 4);
  float* sums_g = alloc((size_t)NE_ * NR_ * D_ * 4);   // reused for meta-path sums
  float* cnt_g  = alloc((size_t)NE_ * NR_ * 4);        // reused for meta-path cnt
  float* feat_a = alloc((size_t)NSUB_ * D_ * 4);
  float* feat_b = alloc((size_t)NSUB_ * D_ * 4);
  float* sem    = alloc((size_t)NMP_ * NREG_ * D_ * 4);
  float* hbuf   = alloc((size_t)NREG_ * D_ * 4);
  float* Ereg   = alloc((size_t)NREG_ * D_ * 4);
  float* attw   = alloc(64 * 4);                       // [0..2]=w_mp, [8..9]=w_task

  auto gemm = [&](const float* A, const float* B, const float* bias, float* C,
                  int M, int N, int K, int accum, int act,
                  const float* cntA, int RA) {
    dim3 grid((M + BM_ - 1) / BM_, N / BN_);
    han_gemm_wmma<<<grid, 128, 0, stream>>>(A, B, bias, C, cntA, RA,
                                            M, N, K, accum, act);
  };

  // 1) Entity mapping: E0 = E_pretrain @ W_map + b_map
  gemm(E_pre, W_map, b_map, E_a, NE_, D_, PRE_, 0, ACT_NONE, nullptr, 0);

  // 2) Global RGCN stack (tanh); mean fused into the einsum GEMM's A staging.
  float* cur = E_a; float* nxt = E_b;
  for (int l = 0; l < SUM_L_; ++l) {
    hipMemsetAsync(sums_g, 0, (size_t)NE_ * NR_ * D_ * 4, stream);
    hipMemsetAsync(cnt_g,  0, (size_t)NE_ * NR_ * 4,      stream);
    {
      int blocks = (EG_ + 7) / 8;   // 8 waves per 256-thread block
      han_scatter<<<blocks, 256, 0, stream>>>(cur, edge_index, edge_index + EG_,
                                              edge_type, sums_g, cnt_g, EG_, NR_);
    }
    // einsum(mean, Wrel) == (sums/cnt)[NE, NR*D] @ [NR*D, D]
    gemm(sums_g, rgcn_Wrel + (size_t)l * NR_ * D_ * D_, rgcn_b + (size_t)l * D_,
         nxt, NE_, D_, NR_ * D_, 0, ACT_NONE, cnt_g, NR_);
    gemm(cur, rgcn_Wroot + (size_t)l * D_ * D_, nullptr,
         nxt, NE_, D_, D_, 1, ACT_TANH, nullptr, 0);
    float* t = cur; cur = nxt; nxt = t;
  }
  // cur == final global embedding E

  // 3) Semantic-attention weights over meta-paths
  han_attn<<<1, 128, 0, stream>>>(mp_emb, Wp1, bp1, Wp2, attw, NMP_, PRE_);

  // 4) HAN: per-meta-path RGCN stacks (relu) on subgraphs
  for (int i = 0; i < NMP_; ++i) {
    {
      int total = NSUB_ * 32;
      han_gather_async<<<(total + 255) / 256, 256, 0, stream>>>(
          cur, mp_eids + (size_t)i * NSUB_, feat_a, NSUB_);
    }
    float* fcur = feat_a; float* fnxt = feat_b;
    const int* s_ptr = mp_ei + (size_t)i * 2 * EM_;
    const int* t_ptr = s_ptr + EM_;
    const int* e_ptr = mp_et + (size_t)i * EM_;
    for (int j = 0; j < SUB_L_; ++j) {
      hipMemsetAsync(sums_g, 0, (size_t)NSUB_ * NRS_ * D_ * 4, stream);
      hipMemsetAsync(cnt_g,  0, (size_t)NSUB_ * NRS_ * 4,      stream);
      {
        int blocks = (EM_ + 7) / 8;
        han_scatter<<<blocks, 256, 0, stream>>>(fcur, s_ptr, t_ptr, e_ptr,
                                                sums_g, cnt_g, EM_, NRS_);
      }
      const size_t wrel_off  = ((size_t)i * SUB_L_ + j) * NRS_ * D_ * D_;
      const size_t wroot_off = ((size_t)i * SUB_L_ + j) * D_ * D_;
      const size_t b_off     = ((size_t)i * SUB_L_ + j) * D_;
      gemm(sums_g, han_Wrel + wrel_off, han_b + b_off, fnxt, NSUB_, D_, NRS_ * D_,
           0, ACT_NONE, cnt_g, NRS_);
      gemm(fcur, han_Wroot + wroot_off, nullptr, fnxt, NSUB_, D_, D_, 1, ACT_RELU,
           nullptr, 0);
      float* t = fcur; fcur = fnxt; fnxt = t;
    }
    // sem_i = feat[:NREG] (contiguous prefix)
    hipMemcpyAsync(sem + (size_t)i * NREG_ * D_, fcur, (size_t)NREG_ * D_ * 4,
                   hipMemcpyDeviceToDevice, stream);
  }

  // 5) h = sum_i w_i * sem_i ; E_reg = h @ Wpred + bpred
  {
    int total4 = NREG_ * 32;
    han_semsum<<<(total4 + 255) / 256, 256, 0, stream>>>(
        (const float4*)(sem + 0),
        (const float4*)(sem + (size_t)1 * NREG_ * D_),
        (const float4*)(sem + (size_t)2 * NREG_ * D_),
        attw, (float4*)hbuf, total4);
  }
  gemm(hbuf, Wpred, bpred, Ereg, NREG_, D_, D_, 0, ACT_NONE, nullptr, 0);

  // 6) Task attention + cross-task fusion + output projection
  han_attn<<<1, 128, 0, stream>>>(task_emb, Wt1, bt1, Wt2, attw + 8, NTASK_, PRE_);
  {
    int blocks = (NREG_ + 7) / 8;
    han_final<<<blocks, 256, 0, stream>>>(
        (const float4*)Ereg, (const float4*)other_reg, (const float4*)other_kg,
        (const float4*)cur, (const float4*)Wer, ber, attw + 8,
        (float*)d_out, NREG_);
  }
}
